// RQVAE_19542101197040
// MI455X (gfx1250) — matrix operand order repaired and checked by reading, not compiled, not run on hardware
//
#include <hip/hip_runtime.h>

// ---------------------------------------------------------------------------
// RQ-VAE forward for MI455X (gfx1250, wave32, WMMA bf16 16x16x32, TDM)
//   B=65536, IN=512, HID=256, LAT=64, L=3, K=2048
// ---------------------------------------------------------------------------

typedef __bf16 bf16_t;
typedef __attribute__((ext_vector_type(16))) __bf16 v16bf;
typedef __attribute__((ext_vector_type(8)))  __bf16 v8bf;
typedef __attribute__((ext_vector_type(8)))  float  v8f;
typedef __attribute__((ext_vector_type(4)))  float  v4f;
typedef __attribute__((ext_vector_type(4)))  unsigned int u32x4;
typedef __attribute__((ext_vector_type(8)))  int i32x8;
typedef __attribute__((ext_vector_type(4)))  int i32x4;

#if defined(__has_builtin)
#if __has_builtin(__builtin_amdgcn_tensor_load_to_lds) && \
    __has_builtin(__builtin_amdgcn_s_wait_tensorcnt)
#define HAVE_TDM 1
#endif
#endif
#ifndef HAVE_TDM
#define HAVE_TDM 0
#endif

#define WMMA_BF16(a, b, c) \
  __builtin_amdgcn_wmma_f32_16x16x32_bf16(false, (a), false, (b), (short)0, (c), false, false)

// raw LDS byte offset of a shared-memory pointer (generic -> AS3 -> int)
__device__ __forceinline__ unsigned lds_offset_of(const void* p) {
  return (unsigned)(unsigned long long)(const __attribute__((address_space(3))) char*)p;
}

// ---- fragment load helpers (contiguous k order: k = 16*half + e) ------------------

// 32 contiguous bf16 bytes -> v16bf  (B fragment: element e <-> k = e + 16*half)
__device__ __forceinline__ v16bf load_bf16x16(const bf16_t* p) {
  v8bf a = *(const v8bf*)p;
  v8bf b = *(const v8bf*)(p + 8);
  v16bf r;
#pragma unroll
  for (int i = 0; i < 8; ++i) { r[i] = a[i]; r[i + 8] = b[i]; }
  return r;
}

// A fragment 16x32 bf16: lane half (lane/16) selects K offset +16.
// element e in [0,16) -> k = k0 + 16*half + e
__device__ __forceinline__ v16bf load_a_bf16(const bf16_t* row_k0, int half) {
  v8bf a = *(const v8bf*)(row_k0 + 16 * half);
  v8bf b = *(const v8bf*)(row_k0 + 16 * half + 8);
  v16bf r;
#pragma unroll
  for (int i = 0; i < 8; ++i) { r[i] = a[i]; r[i + 8] = b[i]; }
  return r;
}

__device__ __forceinline__ v16bf load_a_f32(const float* row_k0, int half) {
  v4f f0 = *(const v4f*)(row_k0 + 16 * half);
  v4f f1 = *(const v4f*)(row_k0 + 16 * half + 4);
  v4f f2 = *(const v4f*)(row_k0 + 16 * half + 8);
  v4f f3 = *(const v4f*)(row_k0 + 16 * half + 12);
  v16bf r;
#pragma unroll
  for (int i = 0; i < 4; ++i) {
    r[i]      = (__bf16)f0[i];
    r[i + 4]  = (__bf16)f1[i];
    r[i + 8]  = (__bf16)f2[i];
    r[i + 12] = (__bf16)f3[i];
  }
  return r;
}

// ---------------------------------------------------------------------------
// Tiled WMMA GEMM:  out = act(A[M,K] @ W[K,N] + bias)
//   block tile 128x64, 8 waves, each wave: 16 rows x 64 cols, K-step 32.
//   W tile staged via LDS, pre-swizzled into B-fragment layout, double buffered.
// ---------------------------------------------------------------------------
template <bool A_F32, bool RELU, bool OUT_BF16, bool OUT_F32>
__global__ __launch_bounds__(256) void gemm_wmma_kernel(
    const void* __restrict__ Av, const bf16_t* __restrict__ W,
    const float* __restrict__ bias,
    bf16_t* __restrict__ outB, float* __restrict__ outF,
    int N, int K) {
  __shared__ bf16_t ldsB[2][2048];  // 2 x (4 ntiles x 32 lanes x 16 bf16)

  const int tid  = threadIdx.x;
  const int wave = tid >> 5;
  const int lane = tid & 31;
  const int lhalf = lane >> 4;
  const int l16   = lane & 15;
  const long rowBase = (long)blockIdx.y * 128 + wave * 16;
  const int  n0      = blockIdx.x * 64;

  const int kk = tid >> 3;           // 0..31 : K row within tile
  const int n8 = (tid & 7) * 8;      // 8 contiguous N per thread
  const int shalf = kk >> 4;
  const int slot  = (kk & 15);       // element index within lane's 16

  v8f acc[4] = {v8f{}, v8f{}, v8f{}, v8f{}};
  const int nk = K >> 5;

  // stage tile 0 (global -> regs -> swizzled LDS)
  {
    v8bf w0 = *(const v8bf*)(W + (long)kk * N + n0 + n8);
#pragma unroll
    for (int j = 0; j < 8; ++j) {
      int n = n8 + j;
      ldsB[0][(n >> 4) * 512 + (shalf * 16 + (n & 15)) * 16 + slot] = w0[j];
    }
  }
  __syncthreads();

  for (int kt = 0; kt < nk; ++kt) {
    const int buf = kt & 1;
    const int k0  = kt << 5;
    const bool haveNext = (kt + 1) < nk;

    // start fetching next W tile while current tile computes
    v8bf wNext;
    if (haveNext)
      wNext = *(const v8bf*)(W + (long)(k0 + 32 + kk) * N + n0 + n8);

    // A fragment (direct global)
    v16bf a;
    if constexpr (A_F32) {
      const float* Ap = (const float*)Av + (rowBase + l16) * (long)K + k0;
      a = load_a_f32(Ap, lhalf);
      if (haveNext) __builtin_prefetch(Ap + 32, 0, 0);   // global_prefetch_b8
    } else {
      const bf16_t* Ap = (const bf16_t*)Av + (rowBase + l16) * (long)K + k0;
      a = load_a_bf16(Ap, lhalf);
      if (haveNext) __builtin_prefetch(Ap + 32, 0, 0);
    }

#pragma unroll
    for (int nt = 0; nt < 4; ++nt) {
      v16bf b = load_bf16x16(&ldsB[buf][nt * 512 + lane * 16]);  // 2x ds_load_b128
      acc[nt] = WMMA_BF16(a, b, acc[nt]);
    }

    __syncthreads();  // all waves done reading before next tile is stored
    if (haveNext) {
#pragma unroll
      for (int j = 0; j < 8; ++j) {
        int n = n8 + j;
        ldsB[buf ^ 1][(n >> 4) * 512 + (shalf * 16 + (n & 15)) * 16 + slot] = wNext[j];
      }
    }
    __syncthreads();  // tile kt+1 visible before next iteration reads it
  }

  // epilogue: D layout -> row = v + 8*half, col = nt*16 + l16
#pragma unroll
  for (int nt = 0; nt < 4; ++nt) {
    const int col = n0 + nt * 16 + l16;
    const float bv = bias[col];
#pragma unroll
    for (int v = 0; v < 8; ++v) {
      float x = acc[nt][v] + bv;
      if constexpr (RELU) x = fmaxf(x, 0.0f);
      const long idx = (rowBase + v + 8 * lhalf) * (long)N + col;
      if constexpr (OUT_F32)  outF[idx] = x;
      if constexpr (OUT_BF16) outB[idx] = (__bf16)x;
    }
  }
}

// ---------------------------------------------------------------------------
// Residual quantization argmin:  code[b] = argmin_k (||C_k||^2 - 2 r_b . C_k)
//   1024 threads (32 waves = 8/SIMD, one block per WGP due to 256KB LDS).
//   Whole 2048x64 bf16 codebook is DMA'd into LDS once per block via the
//   Tensor Data Mover (tensor_load_to_lds + s_wait_tensorcnt), then each
//   wave scans all codes with 2 independent-accumulator WMMAs per 16-code
//   chunk, fragments read as contiguous ds_load_b128.
// ---------------------------------------------------------------------------
__global__ __launch_bounds__(1024) void rq_argmin_kernel(
    const bf16_t* __restrict__ R,     // [B,64] residual, bf16
    const bf16_t* __restrict__ CB,    // [2048,64] codebook, bf16
    const float* __restrict__ norms,  // [2048]
    int* __restrict__ codesI, float* __restrict__ codesF) {
  extern __shared__ bf16_t ldsCB[];   // 2048*64 bf16 = 256 KB (dynamic)

  const int tid  = threadIdx.x;
  const int wave = tid >> 5;
  const int lane = tid & 31;
  const int lhalf = lane >> 4;
  const int l16   = lane & 15;
  const long rowBase = (long)blockIdx.x * 512 + wave * 16;

#if HAVE_TDM
  if (tid < 32) {
    // D# group 0: count=1 | lds_addr | global_addr(57b) | type=2
    const unsigned long long ga = (unsigned long long)(const void*)CB;
    u32x4 g0;
    g0[0] = 1u;                                    // count = 1, user mode
    g0[1] = lds_offset_of(ldsCB);                  // lds_addr (bytes)
    g0[2] = (unsigned)ga;                          // global_addr[31:0]
    g0[3] = (unsigned)((ga >> 32) & 0x1FFFFFFull)  // global_addr[56:32]
            | 0x80000000u;                         // type = 2 ("image")
    // D# group 1: 1-D tensor, data_size=8B, 262144B = 32768 units
    //   tensor_dim0 = tile_dim0 = 32768, stride = 32768
    i32x8 g1 = {};
    g1[0] = (3 << 16);                 // data_size = 8 bytes
    g1[1] = (int)0x80000000u;          // tensor_dim0[15:0]=0x8000 in bits[31:16]
    g1[2] = 0;                         // tensor_dim0 hi / tensor_dim1 = 0
    g1[3] = (int)0x80000000u;          // tile_dim0 = 0x8000 in bits[31:16]
    g1[4] = 0;                         // tile_dim1/2 unused
    g1[5] = 32768;                     // tensor_dim0_stride lo32
    g1[6] = 0;
    g1[7] = 0;
    i32x4 z4 = {0, 0, 0, 0};           // groups 2/3 unused (<=2D tensor)
    i32x8 z8 = {};                     // extra group (6-arg toolchain form)
    __builtin_amdgcn_tensor_load_to_lds(g0, g1, z4, z4, z8, 0);
    __builtin_amdgcn_s_wait_tensorcnt(0);
  }
#else
  // fallback: cooperative copy, 1024 threads x 16 x v8bf
#pragma unroll
  for (int it = 0; it < 16; ++it) {
    const int idx = (it * 1024 + tid) * 8;
    *(v8bf*)(ldsCB + idx) = *(const v8bf*)(CB + idx);
  }
#endif
  __syncthreads();

  const bf16_t* rp = R + (rowBase + l16) * 64;
  const v16bf a0 = load_a_bf16(rp, lhalf);       // K dims  0..31
  const v16bf a1 = load_a_bf16(rp + 32, lhalf);  // K dims 32..63

  float best[8];
  int   bidx[8];
#pragma unroll
  for (int v = 0; v < 8; ++v) { best[v] = 3.0e38f; bidx[v] = 0; }

  for (int c0 = 0; c0 < 2048; c0 += 16) {
    const int n = c0 + l16;
    const bf16_t* cp = ldsCB + n * 64;
    // B = C^T tiles: element e <-> dim (e + 16*half) [+32 for second frag]
    v16bf b0 = load_bf16x16(cp + 16 * lhalf);
    v16bf b1 = load_bf16x16(cp + 32 + 16 * lhalf);
    v8f acc0 = v8f{};
    v8f acc1 = v8f{};
    acc0 = WMMA_BF16(a0, b0, acc0);   // independent accumulators -> ILP
    acc1 = WMMA_BF16(a1, b1, acc1);
    const float nrm = norms[n];
#pragma unroll
    for (int v = 0; v < 8; ++v) {
      float s = nrm - 2.0f * (acc0[v] + acc1[v]);
      if (s < best[v]) { best[v] = s; bidx[v] = n; }
    }
  }

  // reduce min/argmin across the 16 lanes of each half-group (wave32)
#pragma unroll
  for (int v = 0; v < 8; ++v) {
    float b = best[v];
    int   i = bidx[v];
#pragma unroll
    for (int off = 8; off >= 1; off >>= 1) {
      float ob = __shfl_xor(b, off, 16);
      int   oi = __shfl_xor(i, off, 16);
      if (ob < b || (ob == b && oi < i)) { b = ob; i = oi; }
    }
    if (l16 == 0) {
      const long row = rowBase + v + 8 * lhalf;
      codesI[row] = i;
      codesF[row] = (float)i;
    }
  }
}

// ---------------------------------------------------------------------------
// Per-layer update:  r_next = r - C[code];  q += C[code]
// ---------------------------------------------------------------------------
__global__ __launch_bounds__(256) void rq_update_kernel(
    const float* __restrict__ rPrev, const int* __restrict__ codes,
    const float* __restrict__ CBf, float* __restrict__ qAcc,
    float* __restrict__ rNextF, bf16_t* __restrict__ rNextB,
    float* __restrict__ qOutF, bf16_t* __restrict__ qOutB, int layer) {
  const long i = (long)blockIdx.x * 256 + threadIdx.x;  // over B*64
  const long row = i >> 6;
  const int  d   = (int)(i & 63);
  const int  c   = codes[row];
  const float cv = CBf[(long)c * 64 + d];
  const float rn = rPrev[i] - cv;
  const float q  = (layer == 0 ? 0.0f : qAcc[i]) + cv;
  qAcc[i] = q;
  if (layer < 2) {
    rNextF[i] = rn;
    rNextB[i] = (__bf16)rn;
  } else {
    qOutF[i] = q;
    qOutB[i] = (__bf16)q;
  }
}

// ---------------------------------------------------------------------------
// Small prep kernels
// ---------------------------------------------------------------------------
__global__ void f32_to_bf16_kernel(const float* __restrict__ in,
                                   bf16_t* __restrict__ out, int n) {
  int i = blockIdx.x * blockDim.x + threadIdx.x;
  if (i < n) out[i] = (__bf16)in[i];
}

__global__ void cb_norms_kernel(const float* __restrict__ cb,
                                float* __restrict__ norms, int n) {
  int i = blockIdx.x * blockDim.x + threadIdx.x;
  if (i < n) {
    const float* p = cb + (long)i * 64;
    float s = 0.0f;
#pragma unroll
    for (int d = 0; d < 64; ++d) s += p[d] * p[d];
    norms[i] = s;
  }
}

// ---------------------------------------------------------------------------
extern "C" void kernel_launch(void* const* d_in, const int* in_sizes, int n_in,
                              void* d_out, int out_size, void* d_ws, size_t ws_size,
                              hipStream_t stream) {
  (void)in_sizes; (void)n_in; (void)out_size; (void)ws_size;

  const float* x      = (const float*)d_in[0];
  const float* enc_w1 = (const float*)d_in[1];
  const float* enc_b1 = (const float*)d_in[2];
  const float* enc_w2 = (const float*)d_in[3];
  const float* enc_b2 = (const float*)d_in[4];
  const float* cbooks = (const float*)d_in[5];
  const float* dec_w1 = (const float*)d_in[6];
  const float* dec_b1 = (const float*)d_in[7];
  const float* dec_w2 = (const float*)d_in[8];
  const float* dec_b2 = (const float*)d_in[9];

  constexpr int B = 65536, IN = 512, HID = 256, LAT = 64, L = 3, K = 2048;

  // d_out: codes[3,B] | x_recon[B,IN] | quantized[B,LAT] | residuals[3,B,LAT]
  float* out       = (float*)d_out;
  float* out_codes = out;
  float* out_xrec  = out + (size_t)L * B;
  float* out_quant = out_xrec + (size_t)B * IN;
  float* out_resid = out_quant + (size_t)B * LAT;

  // workspace carve-out
  char* w = (char*)d_ws;
  auto carve = [&](size_t bytes) -> char* {
    char* p = w;
    w += (bytes + 255) & ~(size_t)255;
    return p;
  };
  bf16_t* wb_e1  = (bf16_t*)carve((size_t)IN * HID * 2);
  bf16_t* wb_e2  = (bf16_t*)carve((size_t)HID * LAT * 2);
  bf16_t* wb_d1  = (bf16_t*)carve((size_t)LAT * HID * 2);
  bf16_t* wb_d2  = (bf16_t*)carve((size_t)HID * IN * 2);
  bf16_t* wb_cb  = (bf16_t*)carve((size_t)L * K * LAT * 2);
  float*  nrm    = (float*)carve((size_t)L * K * 4);
  bf16_t* h1     = (bf16_t*)carve((size_t)B * HID * 2);
  bf16_t* rbf    = (bf16_t*)carve((size_t)B * LAT * 2);
  float*  qf     = (float*)carve((size_t)B * LAT * 4);
  bf16_t* qbf    = (bf16_t*)carve((size_t)B * LAT * 2);
  bf16_t* h2     = (bf16_t*)carve((size_t)B * HID * 2);
  int*    codesI = (int*)carve((size_t)L * B * 4);

  // weight / codebook bf16 conversion + codebook norms (tiny, one-time cost)
  auto cvt = [&](const float* src, bf16_t* dst, int n) {
    f32_to_bf16_kernel<<<(n + 255) / 256, 256, 0, stream>>>(src, dst, n);
  };
  cvt(enc_w1, wb_e1, IN * HID);
  cvt(enc_w2, wb_e2, HID * LAT);
  cvt(dec_w1, wb_d1, LAT * HID);
  cvt(dec_w2, wb_d2, HID * IN);
  cvt(cbooks, wb_cb, L * K * LAT);
  cb_norms_kernel<<<(L * K + 255) / 256, 256, 0, stream>>>(cbooks, nrm, L * K);

  const dim3 blk(256);

  // encoder: h1 = relu(x @ W1 + b1)   [f32 A, bf16 out]
  gemm_wmma_kernel<true, true, true, false>
      <<<dim3(HID / 64, B / 128), blk, 0, stream>>>(x, wb_e1, enc_b1, h1, nullptr, HID, IN);
  // z = h1 @ W2 + b2  -> residuals[0] (f32) + rbf (bf16)
  gemm_wmma_kernel<false, false, true, true>
      <<<dim3(LAT / 64, B / 128), blk, 0, stream>>>(h1, wb_e2, enc_b2, rbf, out_resid, LAT, HID);

  // residual quantization, 3 sequential layers
  const size_t cbLds = (size_t)K * LAT * sizeof(bf16_t);  // 256 KB dynamic LDS
  for (int l = 0; l < L; ++l) {
    rq_argmin_kernel<<<B / 512, dim3(1024), cbLds, stream>>>(
        rbf, wb_cb + (size_t)l * K * LAT, nrm + (size_t)l * K,
        codesI + (size_t)l * B, out_codes + (size_t)l * B);
    rq_update_kernel<<<(B * LAT) / 256, blk, 0, stream>>>(
        out_resid + (size_t)l * B * LAT, codesI + (size_t)l * B,
        cbooks + (size_t)l * K * LAT, qf,
        out_resid + (size_t)(l < 2 ? l + 1 : l) * B * LAT, rbf,
        out_quant, qbf, l);
  }

  // decoder: h2 = relu(q @ dW1 + db1)
  gemm_wmma_kernel<false, true, true, false>
      <<<dim3(HID / 64, B / 128), blk, 0, stream>>>(qbf, wb_d1, dec_b1, h2, nullptr, HID, LAT);
  // x_recon = h2 @ dW2 + db2  (f32 out straight to d_out)
  gemm_wmma_kernel<false, false, false, true>
      <<<dim3(IN / 64, B / 128), blk, 0, stream>>>(h2, wb_d2, dec_b2, nullptr, out_xrec, IN, HID);
}
